// SimpleGNN_24773371363890
// MI455X (gfx1250) — matrix operand order repaired and checked by reading, not compiled, not run on hardware
//
#include <hip/hip_runtime.h>
#include <hip/hip_bf16.h>

typedef __attribute__((ext_vector_type(16))) __bf16 v16bf;
typedef __attribute__((ext_vector_type(8)))  __bf16 v8bf;
typedef __attribute__((ext_vector_type(8)))  float  v8f;
typedef __attribute__((ext_vector_type(4)))  unsigned int u32x4;
typedef __attribute__((ext_vector_type(8)))  unsigned int u32x8;

#define NNODES 50000
#define NCH    128
#define PW     512
#define NEDGE  800000
#define LDSW_STRIDE 17   // 16 floats + 1 pad DWORD (TDM pad) -> conflict-free

__device__ __forceinline__ unsigned short f32_to_bf16_rne(float f) {
    unsigned int u = __float_as_uint(f);
    u += 0x7FFFu + ((u >> 16) & 1u);
    return (unsigned short)(u >> 16);
}

// ---------------------------------------------------------------------------
// Pack node-projection weights into WMMA B-fragment order (bf16).
// Bpack[jt][ks][lane][f]; B-layout (32x16): n = lane&15, k_local = (lane>>4)*16 + f.
// P columns: [0:128)=Wf_dst [128:256)=Wf_src [256:384)=Ws_dst [384:512)=Ws_src
// ---------------------------------------------------------------------------
__global__ void pack_weights_kernel(const float* __restrict__ Wf,
                                    const float* __restrict__ Ws,
                                    unsigned short* __restrict__ Bpack) {
    int id = blockIdx.x * blockDim.x + threadIdx.x;
    if (id >= 32 * 4 * 32 * 16) return;
    int f    = id & 15;
    int lane = (id >> 4) & 31;
    int ks   = (id >> 9) & 3;
    int jt   = id >> 11;
    int n    = lane & 15;
    int k    = ks * 32 + ((lane >> 4) * 16) + f;   // 0..127
    int jcol = jt * 16 + n;                        // 0..511
    int blk  = jcol >> 7;
    int o    = jcol & 127;
    const float* W = (blk < 2) ? Wf : Ws;
    int koff = (blk & 1) ? 128 : 0;
    Bpack[id] = f32_to_bf16_rne(W[o * 272 + koff + k]);
}

// ---------------------------------------------------------------------------
// Elementwise: hbf (bf16) = acc (f32) = maybe-relu(src).  bf16 activations are
// consumed directly as WMMA A fragments (row-major 16-bit A needs plain b128
// loads per ISA 10.9), so no LDS staging in the GEMM.
// ---------------------------------------------------------------------------
__global__ void prep_kernel(const float* src, unsigned short* hbf, float* acc,
                            int relu, int n) {
    int i = blockIdx.x * blockDim.x + threadIdx.x;
    if (i < n) {
        float v = src[i];
        if (relu) v = fmaxf(v, 0.0f);
        hbf[i] = f32_to_bf16_rne(v);
        acc[i] = v;
    }
}

// ---------------------------------------------------------------------------
// Node projection GEMM:  P[N x 512] = hbf[N x 128] @ Bpack (bf16 -> f32)
// Block = 16 rows, 8 waves x 4 col tiles. A fragment = two b128 global loads
// per lane (row-major 16-bit layout); hoisted across the 4 column tiles.
// ---------------------------------------------------------------------------
__global__ __launch_bounds__(256) void node_proj_kernel(
    const unsigned short* __restrict__ hbf,
    const unsigned short* __restrict__ Bpack,
    float* __restrict__ P) {
    const int tid  = threadIdx.x;
    const int lane = tid & 31;
    const int wave = tid >> 5;
    const int row0 = blockIdx.x * 16;

    const int arow = lane & 15;            // A row within tile
    const int koff = (lane >> 4) * 16;     // lane-half K offset
    const unsigned short* aptr = hbf + (size_t)(row0 + arow) * NCH + koff;

    v16bf afrag[4];
    #pragma unroll
    for (int ks = 0; ks < 4; ++ks) {
        v8bf alo = *(const v8bf*)(aptr + ks * 32);
        v8bf ahi = *(const v8bf*)(aptr + ks * 32 + 8);
        afrag[ks] = __builtin_shufflevector(alo, ahi, 0, 1, 2, 3, 4, 5, 6, 7,
                                            8, 9, 10, 11, 12, 13, 14, 15);
    }

    const int n     = lane & 15;
    const int mhalf = lane >> 4;
    for (int q = 0; q < 4; ++q) {
        int jt = wave + q * 8;             // 0..31
        v8f acc = {};
        #pragma unroll
        for (int ks = 0; ks < 4; ++ks) {
            v16bf b = *(const v16bf*)&Bpack[((jt * 4 + ks) * 32 + lane) * 16];
            acc = __builtin_amdgcn_wmma_f32_16x16x32_bf16(
                false, afrag[ks], false, b, (short)0, acc, false, false);
        }
        int jcol0 = jt * 16;
        #pragma unroll
        for (int r = 0; r < 8; ++r) {
            P[(long long)(row0 + mhalf * 8 + r) * PW + jcol0 + n] = acc[r];
        }
    }
}

// ---------------------------------------------------------------------------
// Edge epilogue. Weight blocks (128x16 f32, strided gather W[o*272+256+k])
// are staged into LDS by the Tensor Data Mover: 2D tile tile_dim0=16 (k),
// tile_dim1=128 (o), tensor_dim0_stride=272, with pad (interval 16 DWORDs,
// amount 1 DWORD) so LDS rows are 17 floats -> conflict-free j*17+k access.
// One wave issues both descriptors, waits TENSORcnt, then syncthreads.
// ---------------------------------------------------------------------------
__global__ __launch_bounds__(128) void edge_kernel(
    const float* __restrict__ P,
    const long long* __restrict__ eidx,      // int64 (2 x E): row0=src, row1=dst
    const float* __restrict__ ew,            // E x 16
    const float* __restrict__ Wf, const float* __restrict__ bf,
    const float* __restrict__ Ws, const float* __restrict__ bs,
    float* __restrict__ acc, int E_) {
    __shared__ __align__(16) float lds_w[2 * 128 * LDSW_STRIDE];
    __shared__ float sbf[128];
    __shared__ float sbs[128];
    const int tid = threadIdx.x;

    if ((tid >> 5) == 0) {  // wave 0 issues the TDM descriptors
        unsigned lds_base = (unsigned)(size_t)(void*)lds_w;  // low 32b = LDS offset
        // D# group1: data_size=4B(code 2), pad_enable, pad_interval=16DW(code 3),
        // pad_amount=1DW(code 0); tensor_dim0=16, tensor_dim1=128,
        // tile_dim0=16, tile_dim1=128, tensor_dim0_stride=272.
        u32x8 g1 = { (2u << 16) | (1u << 20) | (3u << 22),
                     16u << 16,          // tensor_dim0
                     128u << 16,         // tensor_dim1 lo (hi=0)
                     16u << 16,          // tile_dim0
                     128u,               // tile_dim1 (tile_dim2=0)
                     272u,               // tensor_dim0_stride lo
                     0u, 0u };
        unsigned long long gaf =
            (unsigned long long)(size_t)Wf + 256ull * 4ull;
        u32x4 g0f = { 1u, lds_base, (unsigned)gaf,
                      (unsigned)((gaf >> 32) & 0x1FFFFFFull) | (2u << 30) };
        asm volatile("tensor_load_to_lds %0, %1" :: "s"(g0f), "s"(g1) : "memory");
        unsigned long long gas =
            (unsigned long long)(size_t)Ws + 256ull * 4ull;
        u32x4 g0s = { 1u, lds_base + 128u * LDSW_STRIDE * 4u, (unsigned)gas,
                      (unsigned)((gas >> 32) & 0x1FFFFFFull) | (2u << 30) };
        asm volatile("tensor_load_to_lds %0, %1" :: "s"(g0s), "s"(g1) : "memory");
        __builtin_amdgcn_s_wait_tensorcnt(0);
    }
    sbf[tid] = bf[tid];
    sbs[tid] = bs[tid];
    __syncthreads();

    const float* wf_l = lds_w;                        // [o*17 + k]
    const float* ws_l = lds_w + 128 * LDSW_STRIDE;
    const int j = tid;
    for (int e = blockIdx.x; e < E_; e += gridDim.x) {
        __builtin_prefetch(&eidx[e + gridDim.x], 0, 0);
        long long src = eidx[e];
        long long dst = eidx[(long long)E_ + e];
        float pf = P[dst * PW + j]       + P[src * PW + 128 + j] + sbf[j];
        float ps = P[dst * PW + 256 + j] + P[src * PW + 384 + j] + sbs[j];
        const float* ewe = &ew[(long long)e * 16];
        #pragma unroll
        for (int k = 0; k < 16; ++k) {
            float w = ewe[k];
            pf = fmaf(w, wf_l[j * LDSW_STRIDE + k], pf);
            ps = fmaf(w, ws_l[j * LDSW_STRIDE + k], ps);
        }
        float sig = 1.0f / (1.0f + __expf(-pf));
        float sp  = fmaxf(ps, 0.0f) + log1pf(__expf(-fabsf(ps)));
        atomicAdd(&acc[dst * NCH + j], sig * sp);
    }
}

__global__ void zero_kernel(float* p, int n) {
    int i = blockIdx.x * blockDim.x + threadIdx.x;
    if (i < n) p[i] = 0.0f;
}

// stride (gridDim*blockDim) is a multiple of 128 -> each thread owns one channel
__global__ __launch_bounds__(256) void pool_kernel(const float* __restrict__ h,
                                                   float* __restrict__ pooled,
                                                   int total) {
    int g = blockIdx.x * blockDim.x + threadIdx.x;
    int stride = gridDim.x * blockDim.x;
    float s = 0.0f;
    for (int i = g; i < total; i += stride) s += h[i];
    atomicAdd(&pooled[g & 127], s);
}

__global__ void final_kernel(const float* __restrict__ pooled,
                             const float* __restrict__ Wlin,
                             const float* __restrict__ blin,
                             float* __restrict__ out) {
    int o = threadIdx.x;
    if (o < 64) {
        float s = 0.0f;
        for (int k = 0; k < NCH; ++k) s = fmaf(pooled[k], Wlin[o * NCH + k], s);
        out[o] = blin[o] + s * (1.0f / (float)NNODES);
    }
}

extern "C" void kernel_launch(void* const* d_in, const int* in_sizes, int n_in,
                              void* d_out, int out_size, void* d_ws, size_t ws_size,
                              hipStream_t stream) {
    const float*     x    = (const float*)d_in[0];
    const long long* eidx = (const long long*)d_in[1];   // int64
    const float*     ew   = (const float*)d_in[2];
    const float* Wf[3] = { (const float*)d_in[3],  (const float*)d_in[7],  (const float*)d_in[11] };
    const float* bf[3] = { (const float*)d_in[4],  (const float*)d_in[8],  (const float*)d_in[12] };
    const float* Ws[3] = { (const float*)d_in[5],  (const float*)d_in[9],  (const float*)d_in[13] };
    const float* bs[3] = { (const float*)d_in[6],  (const float*)d_in[10], (const float*)d_in[14] };
    const float* Wlin  = (const float*)d_in[15];
    const float* blin  = (const float*)d_in[16];
    float* out = (float*)d_out;

    // Workspace layout (base 256B aligned; every offset stays 32B aligned)
    unsigned short* hbf  = (unsigned short*)d_ws;                 // N*128 bf16
    float* bufB          = (float*)(hbf + (size_t)NNODES * NCH);  // N*128 f32
    float* P             = bufB + (size_t)NNODES * NCH;           // N*512 f32
    unsigned short* Bpack = (unsigned short*)(P + (size_t)NNODES * PW);
    float* pooled        = (float*)(Bpack + 32 * 4 * 32 * 16);

    const int nElem = NNODES * NCH;
    const int ewBlocks = (nElem + 255) / 256;

    const float* layerSrc = x;
    for (int l = 0; l < 3; ++l) {
        int relu = (l > 0) ? 1 : 0;
        prep_kernel<<<ewBlocks, 256, 0, stream>>>(layerSrc, hbf, bufB, relu, nElem);
        pack_weights_kernel<<<256, 256, 0, stream>>>(Wf[l], Ws[l], Bpack);
        node_proj_kernel<<<NNODES / 16, 256, 0, stream>>>(hbf, Bpack, P);
        edge_kernel<<<8192, 128, 0, stream>>>(P, eidx, ew, Wf[l], bf[l],
                                              Ws[l], bs[l], bufB, NEDGE);
        layerSrc = bufB;
    }

    zero_kernel<<<1, 128, 0, stream>>>(pooled, 128);
    pool_kernel<<<500, 256, 0, stream>>>(bufB, pooled, nElem);
    final_kernel<<<1, 64, 0, stream>>>(pooled, Wlin, blin, out);
}
